// SRkNNAttentionMIL__781684048667
// MI455X (gfx1250) — matrix operand order repaired and checked by reading, not compiled, not run on hardware
//
#include <hip/hip_runtime.h>

// ---------------------------------------------------------------------------
// SRkNNAttentionMIL for MI455X (gfx1250, wave32, WMMA bf16)
//   B=2 bags, N=4096 patches, L=200 features (padded to 224 = 7*32)
//   out = [ h (B*N*L f32) | attention (B*N*N f32) ]
//   d_out attention region doubles as scratch for the raw score matrix.
//   workspace: Pn bf16 (B*N*224) + Pt bf16 (B*224*N) ~ 7.34 MB
//
// LDS tiles are fragment-major: each lane's 8 packed-K dwords are 2 runs of 4 (at 4h and 8+4h)
// & 16B-aligned (stride 20 dwords -> conflict-free on 64 banks), so fragment
// loads are 2x ds_load_b128 straight into WMMA operand regs. B-fragments are
// software-pipelined (2-deep) so ds loads overlap WMMA execution. k_score's
// A tile staging uses GLOBAL_LOAD_ASYNC_TO_LDS_B128 (ASYNCcnt) when available.
// ---------------------------------------------------------------------------

#define BAGS  2
#define NPTS  4096
#define LFEAT 200
#define LPAD  224          // 7 * 32, K padding for 16x16x32 WMMA
#define TOPKR 100
#define NEGC  (-1e19f)
#define TSTRIDE 20         // dword stride per row/col slot in LDS tiles

typedef __attribute__((ext_vector_type(16))) __bf16        v16bf;
typedef __attribute__((ext_vector_type(8)))  float         v8f;
typedef __attribute__((ext_vector_type(4)))  unsigned int  v4u;

// --- async global->LDS (gfx1250 ASYNCcnt path), compile-safe fallbacks ------
#if defined(__has_builtin)
#if __has_builtin(__builtin_amdgcn_global_load_async_to_lds_b128)
#define HAVE_ASYNC_LDS 1
#endif
#endif
#ifndef HAVE_ASYNC_LDS
#define HAVE_ASYNC_LDS 0
#endif

#if HAVE_ASYNC_LDS
// Param 1 (probed from clang diagnostic): generic 'int __vector(4) *' source.
// Param 2: LDS destination, address_space(3) per the *_load_*_lds family.
typedef int v4i_vs __attribute__((vector_size(16)));
typedef __attribute__((address_space(3))) v4i_vs as3_v4i;
static __device__ inline void async_copy16(const void* g, void* l) {
    // generic LDS pointer low 32 bits == LDS byte offset (aperture mapping)
    as3_v4i* lp = (as3_v4i*)(unsigned int)(size_t)l;
    __builtin_amdgcn_global_load_async_to_lds_b128((v4i_vs*)g, lp, 0, 0);
}
static __device__ inline void async_wait() {
#if defined(__has_builtin) && __has_builtin(__builtin_amdgcn_s_wait_asynccnt)
    __builtin_amdgcn_s_wait_asynccnt(0);
#else
    asm volatile("s_wait_asynccnt 0x0" ::: "memory");
#endif
}
#else
static __device__ inline void async_copy16(const void* g, void* l) {
    *(v4u*)l = *(const v4u*)g;      // synchronous fallback
}
static __device__ inline void async_wait() {}
#endif

union Frag {                  // 16 bf16 = 8 VGPRs; u[i] packs elements (2i, 2i+1)
    v16bf v;
    v4u   q[2];
    unsigned int u[8];
};

static __device__ inline unsigned short f32_to_bf16(float f) {
    unsigned int u = __float_as_uint(f);
    u += 0x7FFFu + ((u >> 16) & 1u);      // round-to-nearest-even
    return (unsigned short)(u >> 16);
}
static __device__ inline unsigned int pack_bf2(float f0, float f1) {
    return (unsigned int)f32_to_bf16(f0) | ((unsigned int)f32_to_bf16(f1) << 16);
}
static __device__ inline unsigned int fkey(float f) {   // monotone uint key
    unsigned int u = __float_as_uint(f);
    return (u & 0x80000000u) ? ~u : (u | 0x80000000u);
}
static __device__ inline v8f zero8() {
    v8f z;
#pragma unroll
    for (int i = 0; i < 8; ++i) z[i] = 0.f;
    return z;
}

// --- WMMA fragment loads (layouts per CDNA5 ISA 7.12.2, wave32) -------------
static __device__ inline void loadA(Frag& a, const unsigned int* At, int row, int lane) {
    const int kb4 = (lane >> 4) * 4;
    const unsigned int* p = At + row * TSTRIDE;
    a.q[0] = *(const v4u*)(p + kb4);        // K pairs {0..7} or {8..15}
    a.q[1] = *(const v4u*)(p + 8 + kb4);    // K pairs {16..23} or {24..31}
}
static __device__ inline void loadB(Frag& b, const unsigned int* Bs, int col, int lane) {
    const unsigned int* p = Bs + col * TSTRIDE + (lane >> 4) * 4;
    b.q[0] = *(const v4u*)(p);
    b.q[1] = *(const v4u*)(p + 8);
}
static __device__ inline v8f wmma_bf16(const Frag& a, const Frag& b, v8f c) {
    return __builtin_amdgcn_wmma_f32_16x16x32_bf16(false, a.v, false, b.v,
                                                   (short)0, c, false, false);
}

// ---------------------------------------------------------------------------
// Kernel 1: x_proj = x @ W^T  (M=8192, N=224, K=224-padded). Writes bf16 x_proj
// in row-major (Pn[b][n][d]) and transposed (Pt[b][d][n]) layouts.
// ---------------------------------------------------------------------------
__global__ __launch_bounds__(256) void k_proj(const float* __restrict__ x,
                                              const float* __restrict__ W,
                                              unsigned short* __restrict__ Pn,
                                              unsigned short* __restrict__ Pt) {
    __shared__ unsigned int At[128 * TSTRIDE];
    __shared__ unsigned int Bs[LPAD * TSTRIDE];
    const int tid  = threadIdx.x;
    const int wave = tid >> 5, lane = tid & 31;
    const int cl   = lane & 15;
    const int m0   = blockIdx.x * 128;

    v8f acc[14];
#pragma unroll
    for (int j = 0; j < 14; ++j) acc[j] = zero8();

    for (int k0 = 0; k0 < LPAD; k0 += 32) {
        for (int idx = tid; idx < 128 * 16; idx += 256) {
            int r = idx >> 4, c = idx & 15;
            int m = m0 + r, l = k0 + 2 * c;
            float f0 = (l     < LFEAT) ? x[(size_t)m * LFEAT + l]     : 0.f;
            float f1 = (l + 1 < LFEAT) ? x[(size_t)m * LFEAT + l + 1] : 0.f;
            At[r * TSTRIDE + c] = pack_bf2(f0, f1);
        }
        for (int idx = tid; idx < 16 * LPAD; idx += 256) {
            int p = idx & 15, o = idx >> 4;           // p fastest: coalesced W rows
            int l = k0 + 2 * p;
            float f0 = (o < LFEAT && l     < LFEAT) ? W[(size_t)o * LFEAT + l]     : 0.f;
            float f1 = (o < LFEAT && l + 1 < LFEAT) ? W[(size_t)o * LFEAT + l + 1] : 0.f;
            Bs[o * TSTRIDE + p] = pack_bf2(f0, f1);
        }
        __syncthreads();
        Frag a, bb[2];
        loadA(a, At, 16 * wave + cl, lane);
        loadB(bb[0], Bs, cl, lane);
#pragma unroll
        for (int j = 0; j < 14; ++j) {                // pipelined B fragments
            if (j + 1 < 14) loadB(bb[(j + 1) & 1], Bs, 16 * (j + 1) + cl, lane);
            acc[j] = wmma_bf16(a, bb[j & 1], acc[j]);
        }
        __syncthreads();
    }
    const int rowOff = (lane >> 4) * 8;
#pragma unroll
    for (int j = 0; j < 14; ++j) {
        int o = 16 * j + cl;
#pragma unroll
        for (int r = 0; r < 8; ++r) {
            int m = m0 + 16 * wave + rowOff + r;
            unsigned short hv = f32_to_bf16(acc[j][r]);
            Pn[(size_t)m * LPAD + o] = hv;
            int b = m >> 12, n = m & (NPTS - 1);
            Pt[((size_t)b * LPAD + o) * NPTS + n] = hv;
        }
    }
}

// ---------------------------------------------------------------------------
// Kernel 2: score = P @ P^T per bag (4096x4096, K=224). 128x128 tile / WG,
// 8 waves as 2x4, each wave a 64x32 sub-tile (4x2 fragments).
// Double-buffered LDS; A tile staged via async global->LDS b128 copies.
// ---------------------------------------------------------------------------
__global__ __launch_bounds__(256) void k_score(const unsigned short* __restrict__ Pn,
                                               const unsigned short* __restrict__ Pt,
                                               float* __restrict__ score) {
    __shared__ unsigned int At[2][128 * TSTRIDE];
    __shared__ unsigned int Bs[2][128 * TSTRIDE];
    const int tid  = threadIdx.x;
    const int wave = tid >> 5, lane = tid & 31;
    const int cl   = lane & 15;
    const int b  = blockIdx.z;
    const int m0 = blockIdx.y * 128, n0 = blockIdx.x * 128;
    const int wm = wave >> 2, wn = wave & 3;

    v8f acc[4][2];
#pragma unroll
    for (int mi = 0; mi < 4; ++mi)
#pragma unroll
        for (int ni = 0; ni < 2; ++ni) acc[mi][ni] = zero8();

    auto stage = [&](unsigned int* dA, unsigned int* dB, int k0) {
        // A tile: pure 16B copies -> async global->LDS (ASYNCcnt)
        for (int idx = tid; idx < 512; idx += 256) {
            int r = idx >> 2, q = (idx & 3) * 4;      // q = dword quad within row
            const unsigned short* g =
                &Pn[((size_t)(b * NPTS + m0 + r)) * LPAD + k0 + 2 * q];
            async_copy16(g, &dA[r * TSTRIDE + q]);
        }
        // B tile: needs K-pair packing -> synchronous
        for (int idx = tid; idx < 16 * 128; idx += 256) {
            int p = idx >> 7, nn = idx & 127;
            size_t base = ((size_t)b * LPAD + k0 + 2 * p) * NPTS + n0 + nn;
            dB[nn * TSTRIDE + p] =
                (unsigned int)Pt[base] | ((unsigned int)Pt[base + NPTS] << 16);
        }
    };

    stage(At[0], Bs[0], 0);
    async_wait();
    __syncthreads();
    const int NSTEP = LPAD / 32;
    for (int step = 0; step < NSTEP; ++step) {
        const int cur = step & 1;
        if (step + 1 < NSTEP) stage(At[cur ^ 1], Bs[cur ^ 1], (step + 1) * 32);

        Frag a[4], bb[2];
        loadA(a[0], At[cur], 64 * wm + cl, lane);     // first WMMA's operands first
        loadB(bb[0], Bs[cur], 32 * wn + cl, lane);
        loadB(bb[1], Bs[cur], 32 * wn + 16 + cl, lane);
#pragma unroll
        for (int mi = 1; mi < 4; ++mi)
            loadA(a[mi], At[cur], 64 * wm + 16 * mi + cl, lane);
#pragma unroll
        for (int mi = 0; mi < 4; ++mi)
#pragma unroll
            for (int ni = 0; ni < 2; ++ni)
                acc[mi][ni] = wmma_bf16(a[mi], bb[ni], acc[mi][ni]);

        async_wait();                                  // overlap async A copy with WMMAs
        __syncthreads();
    }
    const int rowOff = (lane >> 4) * 8;
#pragma unroll
    for (int mi = 0; mi < 4; ++mi)
#pragma unroll
        for (int ni = 0; ni < 2; ++ni) {
            int n = n0 + 32 * wn + 16 * ni + cl;
#pragma unroll
            for (int r = 0; r < 8; ++r) {
                int m = m0 + 64 * wm + 16 * mi + rowOff + r;
                score[((size_t)b * NPTS + m) * NPTS + n] = acc[mi][ni][r];
            }
        }
}

// ---------------------------------------------------------------------------
// Kernel 3 (fused top-k + softmax): per row, 100th-largest via 32-step bitwise
// radix select over monotone uint keys in LDS; then in-place masked softmax.
// ---------------------------------------------------------------------------
__global__ __launch_bounds__(256) void k_rknn(float* __restrict__ score) {
    __shared__ unsigned int keys[NPTS];
    __shared__ float vals[NPTS];
    __shared__ float red[256];
    __shared__ int cnt;
    const int row = blockIdx.x, tid = threadIdx.x;
    const int n = row & (NPTS - 1);
    float* sr = score + (size_t)row * NPTS;

    for (int j = tid; j < NPTS; j += 256) {
        float s = sr[j];
        vals[j] = s;
        keys[j] = fkey(s);
    }
    __syncthreads();

    unsigned int res = 0;
    for (int bit = 31; bit >= 0; --bit) {
        unsigned int cand = res | (1u << bit);
        if (tid == 0) cnt = 0;
        __syncthreads();
        int local = 0;
        for (int j = tid; j < NPTS; j += 256) local += (keys[j] >= cand) ? 1 : 0;
        atomicAdd(&cnt, local);
        __syncthreads();
        if (cnt >= TOPKR) res = cand;
        __syncthreads();
    }

    float lmax = -3.4e38f;
    for (int j = tid; j < NPTS; j += 256) {
        float s = vals[j];
        bool adj = (j == n) || (keys[j] >= res);
        float m = adj ? s : NEGC * s;
        vals[j] = m;
        lmax = fmaxf(lmax, m);
    }
    red[tid] = lmax;
    __syncthreads();
    for (int s2 = 128; s2 > 0; s2 >>= 1) {
        if (tid < s2) red[tid] = fmaxf(red[tid], red[tid + s2]);
        __syncthreads();
    }
    const float M = red[0];
    __syncthreads();

    float lsum = 0.f;
    for (int j = tid; j < NPTS; j += 256) {
        float e = __expf(vals[j] - M);
        vals[j] = e;
        lsum += e;
    }
    red[tid] = lsum;
    __syncthreads();
    for (int s2 = 128; s2 > 0; s2 >>= 1) {
        if (tid < s2) red[tid] += red[tid + s2];
        __syncthreads();
    }
    const float inv = 1.0f / red[0];
    __syncthreads();
    for (int j = tid; j < NPTS; j += 256) sr[j] = vals[j] * inv;
}

// ---------------------------------------------------------------------------
// Kernel 4: h = attention @ x_proj  (per bag: 4096x4096 @ 4096x224->200).
// attention converted f32->bf16 while staging; double-buffered; 128 K-steps;
// B fragments software-pipelined (2-deep) to overlap ds loads with WMMAs.
// ---------------------------------------------------------------------------
__global__ __launch_bounds__(256) void k_h(const float* __restrict__ att,
                                           const unsigned short* __restrict__ Pn,
                                           float* __restrict__ hout) {
    __shared__ unsigned int At[2][128 * TSTRIDE];
    __shared__ unsigned int Bs[2][LPAD * TSTRIDE];
    const int tid  = threadIdx.x;
    const int wave = tid >> 5, lane = tid & 31;
    const int cl   = lane & 15;
    const int m0   = blockIdx.x * 128;
    const int b    = m0 >> 12;

    v8f acc[14];
#pragma unroll
    for (int j = 0; j < 14; ++j) acc[j] = zero8();

    auto stage = [&](unsigned int* dA, unsigned int* dB, int k0) {
        for (int idx = tid; idx < 128 * 16; idx += 256) {
            int r = idx >> 4, c = idx & 15;
            const float* src = &att[(size_t)(m0 + r) * NPTS + k0 + 2 * c];
            dA[r * TSTRIDE + c] = pack_bf2(src[0], src[1]);
            __builtin_prefetch(src + 64, 0, 0);       // two k-steps ahead
        }
        for (int idx = tid; idx < 16 * LPAD; idx += 256) {
            int p = idx / LPAD, d = idx % LPAD;       // d fastest: coalesced Pn rows
            size_t base = ((size_t)b * NPTS + k0 + 2 * p) * LPAD + d;
            dB[d * TSTRIDE + p] =
                (unsigned int)Pn[base] | ((unsigned int)Pn[base + LPAD] << 16);
        }
    };

    stage(At[0], Bs[0], 0);
    __syncthreads();
    const int NSTEP = NPTS / 32;
    for (int step = 0; step < NSTEP; ++step) {
        const int cur = step & 1;
        if (step + 1 < NSTEP) stage(At[cur ^ 1], Bs[cur ^ 1], (step + 1) * 32);

        Frag a, bb[2];
        loadA(a, At[cur], 16 * wave + cl, lane);
        loadB(bb[0], Bs[cur], cl, lane);
#pragma unroll
        for (int j = 0; j < 14; ++j) {                // pipelined B fragments
            if (j + 1 < 14) loadB(bb[(j + 1) & 1], Bs[cur], 16 * (j + 1) + cl, lane);
            acc[j] = wmma_bf16(a, bb[j & 1], acc[j]);
        }
        __syncthreads();
    }
    const int rowOff = (lane >> 4) * 8;
#pragma unroll
    for (int j = 0; j < 14; ++j) {
        int o = 16 * j + cl;
        if (o < LFEAT) {
#pragma unroll
            for (int r = 0; r < 8; ++r) {
                int m = m0 + 16 * wave + rowOff + r;
                hout[(size_t)m * LFEAT + o] = acc[j][r];
            }
        }
    }
}

// ---------------------------------------------------------------------------
extern "C" void kernel_launch(void* const* d_in, const int* in_sizes, int n_in,
                              void* d_out, int out_size, void* d_ws, size_t ws_size,
                              hipStream_t stream) {
    const float* x = (const float*)d_in[0];   // [2][4096][200] f32
    const float* W = (const float*)d_in[1];   // [200][200] f32

    float* out   = (float*)d_out;
    float* hout  = out;                                        // B*N*L
    float* score = out + (size_t)BAGS * NPTS * LFEAT;          // B*N*N (score -> attention in place)

    unsigned short* Pn = (unsigned short*)d_ws;                // bf16 x_proj row-major
    unsigned short* Pt = Pn + (size_t)BAGS * NPTS * LPAD;      // bf16 x_proj transposed

    k_proj <<<dim3(BAGS * NPTS / 128), 256, 0, stream>>>(x, W, Pn, Pt);
    k_score<<<dim3(NPTS / 128, NPTS / 128, BAGS), 256, 0, stream>>>(Pn, Pt, score);
    k_rknn <<<dim3(BAGS * NPTS), 256, 0, stream>>>(score);
    k_h    <<<dim3(BAGS * NPTS / 128), 256, 0, stream>>>(score, Pn, hout);
}
